// GATBase_25598005085056
// MI455X (gfx1250) — matrix operand order repaired and checked by reading, not compiled, not run on hardware
//
#include <hip/hip_runtime.h>
#include <cmath>

typedef __bf16 bf16_t;
typedef __attribute__((ext_vector_type(16))) __bf16 v16bf;
typedef __attribute__((ext_vector_type(8)))  __bf16 v8bf;
typedef __attribute__((ext_vector_type(8)))  float  v8f;

static __device__ __forceinline__ bf16_t to_bf16(float f) {
  union { float f; unsigned int u; } v; v.f = f;
  unsigned int r = v.u + 0x7FFFu + ((v.u >> 16) & 1u);
  unsigned short h = (unsigned short)(r >> 16);
  bf16_t out; __builtin_memcpy(&out, &h, 2);
  return out;
}

// ---- WMMA fragment helpers (wave32, 16x16x32 bf16) ----
// A: 16x32 bf16, row-major source [rows, lda]
static __device__ __forceinline__ v16bf load_a_frag(const bf16_t* __restrict__ A,
                                                    long row0, int lda, int k0,
                                                    long nrows) {
  int lane = threadIdx.x & 31;
  long m = row0 + (lane & 15);
  if (m >= nrows) m = nrows - 1;
  int kb = (lane >> 4) * 8;            // lanes 0-15: K 0..7/16..23; 16-31: 8..15/24..31
  const bf16_t* p = A + m * (long)lda + k0 + kb;
  v8bf lo = *(const v8bf*)(p);
  v8bf hi = *(const v8bf*)(p + 16);
  return __builtin_shufflevector(lo, hi, 0,1,2,3,4,5,6,7,8,9,10,11,12,13,14,15);
}
// B: 32x16 bf16 from pre-transposed weights Wt[Ncols][K] row-major
static __device__ __forceinline__ v16bf load_b_frag(const bf16_t* __restrict__ Wt,
                                                    int col0, int K, int k0) {
  int lane = threadIdx.x & 31;
  int n = col0 + (lane & 15);
  int kb = (lane >> 4) * 8;            // lanes 0-15: K 0..7/16..23; 16-31: 8..15/24..31
  const bf16_t* p = Wt + (long)n * K + k0 + kb;
  v8bf lo = *(const v8bf*)(p);
  v8bf hi = *(const v8bf*)(p + 16);
  return __builtin_shufflevector(lo, hi, 0,1,2,3,4,5,6,7,8,9,10,11,12,13,14,15);
}

static __device__ __forceinline__ void atomicMaxFloat(float* addr, float val) {
  if (val >= 0.f) atomicMax((int*)addr, __float_as_int(val));
  else            atomicMin((unsigned int*)addr, __float_as_uint(val));
}

// ---------------- utility kernels ----------------
__global__ void k_fill(float* p, float v, long n) {
  long i = (long)blockIdx.x * blockDim.x + threadIdx.x;
  if (i < n) p[i] = v;
}
__global__ void k_copy(float* d, const float* s, long n) {
  long i = (long)blockIdx.x * blockDim.x + threadIdx.x;
  if (i < n) d[i] = s[i];
}
__global__ void k_cast_relu(const float* in, bf16_t* out, long n, int relu) {
  long i = (long)blockIdx.x * blockDim.x + threadIdx.x;
  if (i >= n) return;
  float v = in[i];
  if (relu) v = fmaxf(v, 0.f);
  out[i] = to_bf16(v);
}
// W[k, n] (row-major, K x Nc) -> Wt[n, k] bf16
__global__ void k_transpose_cast(const float* W, bf16_t* Wt, int K, int Nc) {
  int i = blockIdx.x * blockDim.x + threadIdx.x;
  if (i >= K * Nc) return;
  int n = i / K, k = i - n * K;
  Wt[(long)n * K + k] = to_bf16(W[(long)k * Nc + n]);
}
// w_att_e[d,h] = sum_c W_edge[d, h*64+c] * att_edge[h,c]   (one 256-thread block)
__global__ void k_watt(const float* We, const float* ae, float* watt) {
  int t = threadIdx.x;                 // 256 = 64*4
  int d = t >> 2, h = t & 3;
  float s = 0.f;
  const float* wr = We + (long)d * 256 + h * 64;
  const float* ar = ae + h * 64;
  for (int c = 0; c < 64; ++c) s += wr[c] * ar[c];
  watt[d * 4 + h] = s;
}

// ---------------- node projection GEMM: X[N,256] = node_bf @ W_lin ----------------
__global__ void k_node_gemm(const bf16_t* __restrict__ A, const bf16_t* __restrict__ Wt,
                            float* __restrict__ X, long N) {
  long row0 = (long)blockIdx.x * 16;
  int wave = threadIdx.x >> 5;         // 8 waves
  int lane = threadIdx.x & 31;
  // A fragments shared by both column tiles of this wave
  v16bf a0 = load_a_frag(A, row0, 64, 0, N);
  v16bf a1 = load_a_frag(A, row0, 64, 32, N);
  bool full = (row0 + 16 <= N);        // uniform per block
  for (int ct = wave; ct < 16; ct += 8) {
    v8f acc = {};
    v16bf b0 = load_b_frag(Wt, ct * 16, 64, 0);
    v16bf b1 = load_b_frag(Wt, ct * 16, 64, 32);
    acc = __builtin_amdgcn_wmma_f32_16x16x32_bf16(false, a0, false, b0,
                                                  (short)0, acc, false, false);
    acc = __builtin_amdgcn_wmma_f32_16x16x32_bf16(false, a1, false, b1,
                                                  (short)0, acc, false, false);
    int col = ct * 16 + (lane & 15);
    long rb = row0 + (lane >> 4) * 8;
    float* out = X + rb * 256 + col;
    if (full) {
      #pragma unroll
      for (int v = 0; v < 8; ++v) out[(long)v * 256] = acc[v];
    } else {
      for (int v = 0; v < 8; ++v)
        if (rb + v < N) out[(long)v * 256] = acc[v];
    }
  }
}

// s_src/s_dst[n,h] = sum_c X[n,h*64+c] * att_{src,dst}[h,c]
__global__ void k_satt(const float* X, const float* as, const float* ad,
                       float* ssrc, float* sdst, long N) {
  long i = (long)blockIdx.x * blockDim.x + threadIdx.x;
  if (i >= N * 4) return;
  long n = i >> 2; int h = i & 3;
  const float* xr = X + n * 256 + h * 64;
  const float* a0 = as + h * 64;
  const float* a1 = ad + h * 64;
  float s0 = 0.f, s1 = 0.f;
  for (int c = 0; c < 64; ++c) { float v = xr[c]; s0 += v * a0[c]; s1 += v * a1[c]; }
  ssrc[i] = s0; sdst[i] = s1;
}

// alpha[e,h] = edge_feats[e,:] @ w_att_e[:,h]   (one wave per edge)
__global__ void k_alpha_e(const float* ef, const float* watt, float* alpha, long E) {
  long wv = ((long)blockIdx.x * blockDim.x + threadIdx.x) >> 5;
  int lane = threadIdx.x & 31;
  if (wv >= E) return;
  const float* row = ef + wv * 64;
  float a0 = 0.f, a1 = 0.f, a2 = 0.f, a3 = 0.f;
  for (int c = lane; c < 64; c += 32) {
    float v = row[c];
    a0 += v * watt[c * 4 + 0]; a1 += v * watt[c * 4 + 1];
    a2 += v * watt[c * 4 + 2]; a3 += v * watt[c * 4 + 3];
  }
  for (int off = 16; off > 0; off >>= 1) {
    a0 += __shfl_down(a0, off, 32); a1 += __shfl_down(a1, off, 32);
    a2 += __shfl_down(a2, off, 32); a3 += __shfl_down(a3, off, 32);
  }
  if (lane == 0) {
    float* o = alpha + wv * 4;
    o[0] = a0; o[1] = a1; o[2] = a2; o[3] = a3;
  }
}

// leaky-relu logits + segment max (atomic)
__global__ void k_att1(float* alpha, const float* ssrc, const float* sdst,
                       const int* src, const int* dst, float* amax, long E) {
  long i = (long)blockIdx.x * blockDim.x + threadIdx.x;
  if (i >= E * 4) return;
  long e = i >> 2; int h = i & 3;
  int si = src[e], di = dst[e];
  float a = alpha[i] + ssrc[(long)si * 4 + h] + sdst[(long)di * 4 + h];
  a = (a > 0.f) ? a : 0.2f * a;
  alpha[i] = a;
  atomicMaxFloat(&amax[(long)di * 4 + h], a);
}
// exp(a - amax[dst]) + segment sum
__global__ void k_att2(float* alpha, const float* amax, const int* dst,
                       float* denom, long E) {
  long i = (long)blockIdx.x * blockDim.x + threadIdx.x;
  if (i >= E * 4) return;
  long e = i >> 2; int h = i & 3;
  int di = dst[e];
  float a = __expf(alpha[i] - amax[(long)di * 4 + h]);
  alpha[i] = a;
  atomicAdd(&denom[(long)di * 4 + h], a);
}
// agg[dst,c] += (1/H) * sum_h (alpha/denom) * X[src, h*64+c]
__global__ void k_att3(const float* alpha, const float* denom, const float* X,
                       const int* src, const int* dst, float* agg, long E) {
  long i = (long)blockIdx.x * blockDim.x + threadIdx.x;
  if (i >= E * 64) return;
  long e = i >> 6; int c = i & 63;
  int si = src[e], di = dst[e];
  const float* al = alpha + e * 4;
  const float* dn = denom + (long)di * 4;
  const float* xr = X + (long)si * 256 + c;
  float s = 0.f;
  for (int h = 0; h < 4; ++h)
    s += (al[h] / (dn[h] + 1e-16f)) * xr[h * 64];
  atomicAdd(&agg[(long)di * 64 + c], s * 0.25f);
}

// per-node: +bias, LayerNorm(64) -> node_ln      (block = 64 threads = 1 node)
__global__ void k_node_post(const float* agg, const float* bias, const float* g,
                            const float* b, float* out) {
  __shared__ float s[64], sq[64];
  long n = blockIdx.x; int t = threadIdx.x;
  float v = agg[n * 64 + t] + bias[t];
  s[t] = v; sq[t] = v * v;
  __syncthreads();
  for (int off = 32; off > 0; off >>= 1) {
    if (t < off) { s[t] += s[t + off]; sq[t] += sq[t + off]; }
    __syncthreads();
  }
  float mu = s[0] * (1.f / 64.f);
  float var = sq[0] * (1.f / 64.f) - mu * mu;
  float r = rsqrtf(var + 1e-5f);
  out[n * 64 + t] = (v - mu) * r * g[t] + b[t];
}

// build h_in = [ni+nj, |ni-nj|, edge] in bf16
__global__ void k_hin(const float* nl, const float* ef, const int* src,
                      const int* dst, bf16_t* Hin, long E) {
  long i = (long)blockIdx.x * blockDim.x + threadIdx.x;
  if (i >= E * 64) return;
  long e = i >> 6; int c = i & 63;
  int si = src[e], di = dst[e];
  float ni = nl[(long)si * 64 + c];
  float nj = nl[(long)di * 64 + c];
  bf16_t* row = Hin + e * 192;
  row[c]       = to_bf16(ni + nj);
  row[64 + c]  = to_bf16(fabsf(ni - nj));
  row[128 + c] = to_bf16(ef[e * 64 + c]);
}

// h1 = relu(LN(Hin @ eW1 + eb1))  -> bf16   (block = 384 threads = 12 waves, 16 edges)
__global__ void k_edge_gemm1(const bf16_t* __restrict__ Hin, const bf16_t* __restrict__ Wt,
                             const float* __restrict__ eb1, const float* __restrict__ g,
                             const float* __restrict__ b, bf16_t* __restrict__ H1, long E) {
  __shared__ float s_sum[16], s_sq[16];
  long row0 = (long)blockIdx.x * 16;
  int wave = threadIdx.x >> 5;         // col tile 0..11
  int lane = threadIdx.x & 31;
  if (threadIdx.x < 16) { s_sum[threadIdx.x] = 0.f; s_sq[threadIdx.x] = 0.f; }
  __syncthreads();
  v8f acc = {};
  #pragma unroll
  for (int k0 = 0; k0 < 192; k0 += 32) {
    v16bf a = load_a_frag(Hin, row0, 192, k0, E);
    v16bf bb = load_b_frag(Wt, wave * 16, 192, k0);
    acc = __builtin_amdgcn_wmma_f32_16x16x32_bf16(false, a, false, bb,
                                                  (short)0, acc, false, false);
  }
  int col = wave * 16 + (lane & 15);
  int rloc = (lane >> 4) * 8;
  float bias = eb1[col];
  float vals[8];
  #pragma unroll
  for (int v = 0; v < 8; ++v) {
    vals[v] = acc[v] + bias;
    atomicAdd(&s_sum[rloc + v], vals[v]);
    atomicAdd(&s_sq[rloc + v], vals[v] * vals[v]);
  }
  __syncthreads();
  float gg = g[col], bbv = b[col];
  bool full = (row0 + 16 <= E);        // uniform per block
  bf16_t* out = H1 + (row0 + rloc) * 192 + col;
  if (full) {
    #pragma unroll
    for (int v = 0; v < 8; ++v) {
      float mu = s_sum[rloc + v] * (1.f / 192.f);
      float var = s_sq[rloc + v] * (1.f / 192.f) - mu * mu;
      float h = (vals[v] - mu) * rsqrtf(var + 1e-5f) * gg + bbv;
      out[(long)v * 192] = to_bf16(fmaxf(h, 0.f));
    }
  } else {
    for (int v = 0; v < 8; ++v) {
      if (row0 + rloc + v >= E) continue;
      float mu = s_sum[rloc + v] * (1.f / 192.f);
      float var = s_sq[rloc + v] * (1.f / 192.f) - mu * mu;
      float h = (vals[v] - mu) * rsqrtf(var + 1e-5f) * gg + bbv;
      out[(long)v * 192] = to_bf16(fmaxf(h, 0.f));
    }
  }
}

// edge_new = H1 @ eW2 + eb2 + edge_old   (block = 128 threads = 4 waves, 16 edges)
__global__ void k_edge_gemm2(const bf16_t* __restrict__ H1, const bf16_t* __restrict__ Wt,
                             const float* __restrict__ eb2, const float* __restrict__ e_old,
                             float* __restrict__ e_new, long E) {
  long row0 = (long)blockIdx.x * 16;
  int wave = threadIdx.x >> 5;         // col tile 0..3
  int lane = threadIdx.x & 31;
  v8f acc = {};
  #pragma unroll
  for (int k0 = 0; k0 < 192; k0 += 32) {
    v16bf a = load_a_frag(H1, row0, 192, k0, E);
    v16bf bb = load_b_frag(Wt, wave * 16, 192, k0);
    acc = __builtin_amdgcn_wmma_f32_16x16x32_bf16(false, a, false, bb,
                                                  (short)0, acc, false, false);
  }
  int col = wave * 16 + (lane & 15);
  long rb = row0 + (lane >> 4) * 8;
  float bias = eb2[col];
  bool full = (row0 + 16 <= E);        // uniform per block
  const float* oldp = e_old + rb * 64 + col;
  float* newp = e_new + rb * 64 + col;
  if (full) {
    #pragma unroll
    for (int v = 0; v < 8; ++v) newp[(long)v * 64] = acc[v] + bias + oldp[(long)v * 64];
  } else {
    for (int v = 0; v < 8; ++v)
      if (rb + v < E) newp[(long)v * 64] = acc[v] + bias + oldp[(long)v * 64];
  }
}

extern "C" void kernel_launch(void* const* d_in, const int* in_sizes, int n_in,
                              void* d_out, int out_size, void* d_ws, size_t ws_size,
                              hipStream_t stream) {
  const float* node_in  = (const float*)d_in[0];
  const float* edge_in  = (const float*)d_in[1];
  const int*   ei       = (const int*)d_in[2];
  const float* W_lin    = (const float*)d_in[3];
  const float* W_edge   = (const float*)d_in[4];
  const float* att_src  = (const float*)d_in[5];
  const float* att_dst  = (const float*)d_in[6];
  const float* att_edge = (const float*)d_in[7];
  const float* bias_cv  = (const float*)d_in[8];
  const float* ln_g     = (const float*)d_in[9];
  const float* ln_b     = (const float*)d_in[10];
  const float* eW1      = (const float*)d_in[11];
  const float* eb1      = (const float*)d_in[12];
  const float* eln_g    = (const float*)d_in[13];
  const float* eln_b    = (const float*)d_in[14];
  const float* eW2      = (const float*)d_in[15];
  const float* eb2      = (const float*)d_in[16];

  const long N = in_sizes[0] / 64;
  const long E = in_sizes[1] / 64;
  const int* src = ei;
  const int* dst = ei + E;

  // ---- workspace carve ----
  char* p = (char*)d_ws;
  auto carve = [&](size_t bytes) -> void* {
    void* r = (void*)p;
    p += (bytes + 255) & ~(size_t)255;
    return r;
  };
  float*  X        = (float*)carve((size_t)N * 256 * 4);
  bf16_t* node_bf  = (bf16_t*)carve((size_t)N * 64 * 2);
  float*  node_ln  = (float*)carve((size_t)N * 64 * 4);
  float*  ssrc     = (float*)carve((size_t)N * 4 * 4);
  float*  sdst     = (float*)carve((size_t)N * 4 * 4);
  float*  amax     = (float*)carve((size_t)N * 4 * 4);
  float*  denom    = (float*)carve((size_t)N * 4 * 4);
  float*  agg      = (float*)carve((size_t)N * 64 * 4);
  float*  alpha    = (float*)carve((size_t)E * 4 * 4);
  bf16_t* Hin      = (bf16_t*)carve((size_t)E * 192 * 2);
  bf16_t* H1       = (bf16_t*)carve((size_t)E * 192 * 2);
  float*  edge_buf = (float*)carve((size_t)E * 64 * 4);
  bf16_t* Wt_lin   = (bf16_t*)carve((size_t)256 * 64 * 2);
  bf16_t* Wt_e1    = (bf16_t*)carve((size_t)192 * 192 * 2);
  bf16_t* Wt_e2    = (bf16_t*)carve((size_t)64 * 192 * 2);
  float*  w_att_e  = (float*)carve((size_t)64 * 4 * 4);

  float* out_node = (float*)d_out;
  float* out_edge = out_node + N * 64;

  auto cdiv = [](long a, long b) { return (a + b - 1) / b; };
  const int T = 256;

  const float* edge_cur = edge_in;
  for (int l = 0; l < 4; ++l) {
    // per-layer weight prep (bf16, transposed for B fragments)
    k_transpose_cast<<<cdiv(64 * 256, T), T, 0, stream>>>(W_lin + (long)l * 64 * 256, Wt_lin, 64, 256);
    k_transpose_cast<<<cdiv(192 * 192, T), T, 0, stream>>>(eW1 + (long)l * 192 * 192, Wt_e1, 192, 192);
    k_transpose_cast<<<cdiv(192 * 64, T), T, 0, stream>>>(eW2 + (long)l * 192 * 64, Wt_e2, 192, 64);
    k_watt<<<1, 256, 0, stream>>>(W_edge + (long)l * 64 * 256, att_edge + (long)l * 256, w_att_e);

    // node feats -> bf16 (relu applied to previous layer's LN output for l>0)
    k_cast_relu<<<cdiv(N * 64, T), T, 0, stream>>>(l == 0 ? node_in : node_ln,
                                                   node_bf, N * 64, l > 0 ? 1 : 0);
    // x = node @ W_lin   (WMMA)
    k_node_gemm<<<cdiv(N, 16), 256, 0, stream>>>(node_bf, Wt_lin, X, N);
    // per-node attention scalars
    k_satt<<<cdiv(N * 4, T), T, 0, stream>>>(X, att_src + (long)l * 256,
                                             att_dst + (long)l * 256, ssrc, sdst, N);
    // per-edge attention logit from factored edge projection
    k_alpha_e<<<cdiv(E * 32, T), T, 0, stream>>>(edge_cur, w_att_e, alpha, E);
    // softmax over incoming edges
    k_fill<<<cdiv(N * 4, T), T, 0, stream>>>(amax, -INFINITY, N * 4);
    k_fill<<<cdiv(N * 4, T), T, 0, stream>>>(denom, 0.f, N * 4);
    k_fill<<<cdiv(N * 64, T), T, 0, stream>>>(agg, 0.f, N * 64);
    k_att1<<<cdiv(E * 4, T), T, 0, stream>>>(alpha, ssrc, sdst, src, dst, amax, E);
    k_att2<<<cdiv(E * 4, T), T, 0, stream>>>(alpha, amax, dst, denom, E);
    k_att3<<<cdiv(E * 64, T), T, 0, stream>>>(alpha, denom, X, src, dst, agg, E);
    // +bias, LN -> node_ln
    k_node_post<<<N, 64, 0, stream>>>(agg, bias_cv + (long)l * 64,
                                      ln_g + (long)l * 64, ln_b + (long)l * 64, node_ln);
    // edge MLP (WMMA, fused bias+LN+relu epilogue)
    k_hin<<<cdiv(E * 64, T), T, 0, stream>>>(node_ln, edge_cur, src, dst, Hin, E);
    k_edge_gemm1<<<cdiv(E, 16), 384, 0, stream>>>(Hin, Wt_e1, eb1 + (long)l * 192,
                                                  eln_g + (long)l * 192,
                                                  eln_b + (long)l * 192, H1, E);
    float* edge_next = (l % 2 == 0) ? edge_buf : out_edge;
    k_edge_gemm2<<<cdiv(E, 16), 128, 0, stream>>>(H1, Wt_e2, eb2 + (long)l * 64,
                                                  edge_cur, edge_next, E);
    edge_cur = edge_next;
  }
  // final node output (post-LN, no relu on last layer)
  k_copy<<<cdiv(N * 64, T), T, 0, stream>>>(out_node, node_ln, N * 64);
}
